// RoIHeadTemplate_10307921511151
// MI455X (gfx1250) — compile-verified, matrix operand order checked by reading, not run on hardware
//
#include <hip/hip_runtime.h>
#include <hip/hip_bf16.h>
#include <math.h>

// ---------------------------------------------------------------------------
// RoIHead NMS pipeline for MI455X (gfx1250, wave32).
//   B=4 batches, N=16384 boxes, PRE=512, POST=128, THR=0.7
// Stages:
//   1) scores = max over 3 classes
//   2) O(N) histogram-threshold top-512 (monotone key, 65536 bins) +
//      exact stable rank among ~550 candidates (was O(N^2) rank count)
//   3) gather boxes + center norms + half-diagonal radii
//   4) WMMA (f16->f32, 16x16x32) Gram matrix of centers -> conservative
//      distance gate (kills ~98% of the 262k pairs per batch)
//   5) exact rotated-rect IoU (Sutherland-Hodgman) on gated pairs only
//   6) sequential greedy NMS + stable top-128 + output gather (1 block/batch)
//   7) gt-box -> roi-local transform
// ---------------------------------------------------------------------------

#define BB   4
#define NN   16384
#define PRE  512
#define POST 128
#define MAXC 4096          // candidate cap for threshold bin overflow
#define NBIN 65536         // key>>16 bins
#define NEGV (-1.0e9f)
#define TWO_PI_F 6.2831853071795864f
#define PI_F     3.1415926535897932f
#define HALF_PI_F 1.5707963267948966f

typedef __attribute__((ext_vector_type(16))) _Float16 v16h;
typedef __attribute__((ext_vector_type(8)))  float    v8f;

// monotone float -> uint key: larger float => larger key
__device__ __forceinline__ unsigned int score_key(float f) {
  unsigned int u = __float_as_uint(f);
  return (u & 0x80000000u) ? ~u : (u | 0x80000000u);
}

// ---------------- Stage 1: per-box score = max over classes ----------------
__global__ void score_kernel(const float* __restrict__ cls,
                             float* __restrict__ scores) {
  int idx = blockIdx.x * blockDim.x + threadIdx.x;       // B*N threads
  const float* c = cls + (size_t)idx * 3;
  scores[idx] = fmaxf(c[0], fmaxf(c[1], c[2]));
}

// ---------------- Stage 2a: zero histogram + counters ----------------------
__global__ void zero_kernel(int* __restrict__ p, int n) {
  int idx = blockIdx.x * blockDim.x + threadIdx.x;
  if (idx < n) p[idx] = 0;
}

// ---------------- Stage 2b: 65536-bin key histogram per batch --------------
__global__ void hist_count_kernel(const float* __restrict__ scores,
                                  int* __restrict__ hist) {
  int idx = blockIdx.x * blockDim.x + threadIdx.x;       // B*N threads
  int b = idx >> 14;
  unsigned int bin = score_key(scores[idx]) >> 16;
  atomicAdd(&hist[b * NBIN + bin], 1);
}

// ---------------- Stage 2c: find threshold bin (descending scan) -----------
// 1024 threads: each sums a 64-bin descending chunk; thread 0 walks chunks.
__global__ __launch_bounds__(1024) void hist_scan_kernel(
    const int* __restrict__ hist, int* __restrict__ thr_bin) {
  int b = blockIdx.x;
  int c = threadIdx.x;                 // chunk id, 0 = highest bins
  __shared__ int part[1024];
  int binHigh = NBIN - 1 - c * 64;
  int s = 0;
#pragma unroll
  for (int k = 0; k < 64; ++k) s += hist[b * NBIN + binHigh - k];
  part[c] = s;
  __syncthreads();
  if (c == 0) {
    int cum = 0, T = 0;
    for (int cc = 0; cc < 1024; ++cc) {
      if (cum + part[cc] >= PRE) {
        int bh = NBIN - 1 - cc * 64;
        for (int k = 0; k < 64; ++k) {
          int t = bh - k;
          int h = hist[b * NBIN + t];
          if (cum + h >= PRE) { T = t; break; }
          cum += h;
        }
        break;
      }
      cum += part[cc];
    }
    thr_bin[b] = T;
  }
}

// ---------------- Stage 2d: compact candidates (bin >= T) ------------------
__global__ void compact_kernel(const float* __restrict__ scores,
                               const int* __restrict__ thr_bin,
                               int* __restrict__ cand_cnt,
                               float* __restrict__ candS,
                               int* __restrict__ candI) {
  int idx = blockIdx.x * blockDim.x + threadIdx.x;       // B*N threads
  int b = idx >> 14;
  int i = idx & (NN - 1);
  float s = scores[idx];
  if ((int)(score_key(s) >> 16) >= thr_bin[b]) {
    int pos = atomicAdd(&cand_cnt[b], 1);
    if (pos < MAXC) {
      candS[b * MAXC + pos] = s;
      candI[b * MAXC + pos] = i;
    }
  }
}

// ---------------- Stage 2e: exact stable rank among candidates -------------
// rank depends only on (score, index) pairs -> deterministic output even
// though compaction order is not.
__global__ __launch_bounds__(256) void cand_rank_kernel(
    const float* __restrict__ candS, const int* __restrict__ candI,
    const int* __restrict__ cand_cnt,
    float* __restrict__ top_s, int* __restrict__ top_i) {
  int b = blockIdx.x;
  int C = cand_cnt[b];
  if (C > MAXC) C = MAXC;
  const float* cs = candS + (size_t)b * MAXC;
  const int* ci = candI + (size_t)b * MAXC;
  __shared__ float ts[256];
  __shared__ int ti[256];
  int outer = ((C + 255) >> 8) << 8;
  for (int i0 = 0; i0 < outer; i0 += 256) {
    int myi = i0 + (int)threadIdx.x;
    bool have = myi < C;
    float msc = have ? cs[myi] : 0.0f;
    int mix = have ? ci[myi] : 0;
    int rank = 0;
    for (int j0 = 0; j0 < C; j0 += 256) {
      __syncthreads();
      int jj = j0 + (int)threadIdx.x;
      ts[threadIdx.x] = (jj < C) ? cs[jj] : NEGV;
      ti[threadIdx.x] = (jj < C) ? ci[jj] : 0x7fffffff;
      __syncthreads();
      int lim = C - j0; if (lim > 256) lim = 256;
      for (int k = 0; k < lim; ++k) {
        float v = ts[k];
        rank += (v > msc) || (v == msc && ti[k] < mix);
      }
    }
    if (have && rank < PRE) {
      top_s[b * PRE + rank] = msc;
      top_i[b * PRE + rank] = mix;
    }
  }
}

// ---------------- Stage 3: gather selected boxes, norms, radii -------------
__global__ void gather_kernel(const float* __restrict__ boxes,
                              const int* __restrict__ top_i,
                              float* __restrict__ boxesG,
                              float* __restrict__ norms,
                              float* __restrict__ rads) {
  int idx = blockIdx.x * blockDim.x + threadIdx.x;       // B*PRE threads
  int b = idx >> 9;
  int j = top_i[idx];
  const float* src = boxes + ((size_t)b * NN + j) * 7;
  float* dst = boxesG + (size_t)idx * 7;
  float x = src[0], y = src[1], dx = src[3], dy = src[4];
#pragma unroll
  for (int c = 0; c < 7; ++c) dst[c] = src[c];
  norms[idx] = x * x + y * y;
  rads[idx]  = 0.5f * sqrtf(dx * dx + dy * dy);
}

// ---------------- Stage 4: WMMA center-distance gate -----------------------
// One wave = one 16x16 tile of the 512x512 pairwise dot-product matrix.
// A (16x32 f16): row M=i0+lane (lanes 0-15), K0=x, K1=y, rest 0.
// B (32x16 f16): col N=j0+lane (lanes 0-15), K0=x, K1=y, rest 0.
// dist^2 = |ci|^2 + |cj|^2 - 2*dot (norms in f32);  margin absorbs f16 error.
__global__ __launch_bounds__(32) void wmma_gate_kernel(
    const float* __restrict__ boxesG,
    const float* __restrict__ norms,
    const float* __restrict__ rads,
    float* __restrict__ gate) {
  int blk = blockIdx.x;                  // BB * 32 * 32 blocks
  int b   = blk >> 10;
  int t   = blk & 1023;
  int i0  = (t >> 5) * 16;
  int j0  = (t & 31) * 16;
  int lane = threadIdx.x;
  int lo = lane & 15;
  float on = (lane < 16) ? 1.0f : 0.0f;  // zero inactive lane payload

  const float* bi = boxesG + ((size_t)b * PRE + (i0 + lo)) * 7;
  const float* bj = boxesG + ((size_t)b * PRE + (j0 + lo)) * 7;
  float axf = bi[0] * on, ayf = bi[1] * on;
  float bxf = bj[0] * on, byf = bj[1] * on;

  v16h a = {};
  v16h bmat = {};
  a[0] = (_Float16)axf;  a[1] = (_Float16)ayf;
  bmat[0] = (_Float16)bxf;  bmat[1] = (_Float16)byf;

  v8f c = {};
  c = __builtin_amdgcn_wmma_f32_16x16x32_f16(
      /*neg_a=*/false, a, /*neg_b=*/false, bmat,
      /*c_mod=*/(short)0, c, /*reuse_a=*/false, /*reuse_b=*/false);

  // D layout: VGPR r -> M=r (lanes 0-15) / M=8+r (lanes 16-31), N = lane&15
  int j  = j0 + lo;
  float nj = norms[b * PRE + j];
  float rj = rads[b * PRE + j];
  int mbase = (lane >= 16) ? 8 : 0;
#pragma unroll
  for (int r = 0; r < 8; ++r) {
    int i = i0 + mbase + r;
    float dist2 = norms[b * PRE + i] + nj - 2.0f * c[r];
    float rr = rads[b * PRE + i] + rj + 2.0f;   // +2.0 m f16 safety margin
    gate[((size_t)b * PRE + i) * PRE + j] = (dist2 <= rr * rr) ? 1.0f : 0.0f;
  }
}

// ---------------- Stage 5: exact rotated-rect IoU on gated pairs -----------
__device__ __forceinline__ void rect_corners(const float* __restrict__ bx,
                                             float* cx, float* cy) {
  float c = cosf(bx[6]), s = sinf(bx[6]);
  float hx = 0.5f * bx[3], hy = 0.5f * bx[4];
  const float lx[4] = { hx, hx, -hx, -hx };
  const float ly[4] = { -hy, hy, hy, -hy };   // CCW
#pragma unroll
  for (int k = 0; k < 4; ++k) {
    cx[k] = lx[k] * c - ly[k] * s + bx[0];
    cy[k] = lx[k] * s + ly[k] * c + bx[1];
  }
}

__device__ float rect_iou(const float* __restrict__ A,
                          const float* __restrict__ Bx) {
  float ax[4], ay[4], bx[4], by[4];
  rect_corners(A, ax, ay);
  rect_corners(Bx, bx, by);
  float px[10], py[10], qx[10], qy[10];
  int n = 4;
#pragma unroll
  for (int k = 0; k < 4; ++k) { px[k] = ax[k]; py[k] = ay[k]; }
  // Clip polygon A by the 4 CCW half-planes of B (Sutherland-Hodgman)
  for (int e = 0; e < 4; ++e) {
    float p0x = bx[e], p0y = by[e];
    float ex = bx[(e + 1) & 3] - p0x, ey = by[(e + 1) & 3] - p0y;
    int m = 0;
    for (int k = 0; k < n; ++k) {
      int kn = (k + 1 == n) ? 0 : k + 1;
      float cx0 = px[k], cy0 = py[k], nx0 = px[kn], ny0 = py[kn];
      float dc = ex * (cy0 - p0y) - ey * (cx0 - p0x);
      float dn = ex * (ny0 - p0y) - ey * (nx0 - p0x);
      if (dc >= 0.0f) { qx[m] = cx0; qy[m] = cy0; ++m; }
      if ((dc >= 0.0f) != (dn >= 0.0f)) {
        float t = dc / (dc - dn);
        qx[m] = cx0 + t * (nx0 - cx0);
        qy[m] = cy0 + t * (ny0 - cy0);
        ++m;
      }
    }
    n = m;
    for (int k = 0; k < n; ++k) { px[k] = qx[k]; py[k] = qy[k]; }
    if (n == 0) break;
  }
  float a2 = 0.0f;
  for (int k = 0; k < n; ++k) {
    int kn = (k + 1 == n) ? 0 : k + 1;
    a2 += px[k] * py[kn] - px[kn] * py[k];
  }
  float inter = 0.5f * fabsf(a2);
  float areaA = A[3] * A[4], areaB = Bx[3] * Bx[4];
  return inter / fmaxf(areaA + areaB - inter, 1e-6f);
}

__global__ void iou_kernel(const float* __restrict__ boxesG,
                           const float* __restrict__ gate,
                           float* __restrict__ iou) {
  int idx = blockIdx.x * blockDim.x + threadIdx.x;  // B*PRE*PRE threads
  int b = idx >> 18;
  int rem = idx & (PRE * PRE - 1);
  int i = rem >> 9, j = rem & (PRE - 1);
  const float* Ab = boxesG + ((size_t)(b * PRE + i)) * 7;
  const float* Bb = boxesG + ((size_t)(b * PRE + j)) * 7;
  // overlap box-row fetch with the gate load (global_prefetch_b8)
  __builtin_prefetch(Ab, 0, 0);
  __builtin_prefetch(Bb, 0, 0);
  float v = 0.0f;
  if (gate[idx] != 0.0f) v = rect_iou(Ab, Bb);
  iou[idx] = v;
}

// ---------------- Stage 6: sequential NMS + top-128 + output gather --------
__global__ __launch_bounds__(PRE) void nms_select_kernel(
    const float* __restrict__ iou,
    const float* __restrict__ top_s,
    const int* __restrict__ top_i,
    const float* __restrict__ boxesG,
    const float* __restrict__ cls,
    float* __restrict__ out) {
  int b = blockIdx.x;
  int tid = threadIdx.x;
  __shared__ int keep[PRE];
  __shared__ float ms[PRE];
  __shared__ int sup;
  keep[tid] = 1;
  __syncthreads();
  const float* iouB = iou + (size_t)b * PRE * PRE;
  for (int i = 1; i < PRE; ++i) {
    if (tid == 0) sup = 0;
    __syncthreads();
    if (tid < i && keep[tid] && iouB[(size_t)i * PRE + tid] > 0.7f)
      atomicOr(&sup, 1);
    __syncthreads();
    if (tid == 0) keep[i] = sup ? 0 : 1;
    __syncthreads();
  }
  float msv = keep[tid] ? top_s[b * PRE + tid] : NEGV;
  ms[tid] = msv;
  __syncthreads();
  int rank = 0;
  for (int k = 0; k < PRE; ++k) {
    float v = ms[k];
    rank += (v > msv) || (v == msv && k < tid);
  }
  if (rank < POST) {
    bool valid = msv > NEGV * 0.5f;
    float* roi = out + ((size_t)(b * POST + rank)) * 7;
    const float* src = boxesG + ((size_t)(b * PRE + tid)) * 7;
#pragma unroll
    for (int c = 0; c < 7; ++c) roi[c] = valid ? src[c] : 0.0f;
    out[BB * POST * 7 + b * POST + rank] = valid ? top_s[b * PRE + tid] : 0.0f;
    int lbl = 0;
    if (valid) {
      int sel = top_i[b * PRE + tid];
      const float* cr = cls + ((size_t)(b * NN + sel)) * 3;
      float mx = cr[0];
      if (cr[1] > mx) { mx = cr[1]; lbl = 1; }
      if (cr[2] > mx) { lbl = 2; }
    }
    out[BB * POST * 7 + BB * POST + b * POST + rank] = (float)(lbl + 1);
  }
}

// ---------------- Stage 7: gt boxes -> roi-local frame ---------------------
__global__ void gtlocal_kernel(const float* __restrict__ gt,
                               float* __restrict__ out) {
  int idx = blockIdx.x * blockDim.x + threadIdx.x;   // B*POST threads
  const float* roi = out + (size_t)idx * 7;          // rois written earlier
  const float* g = gt + (size_t)idx * 8;
  float rry = fmodf(roi[6], TWO_PI_F);
  if (rry < 0.0f) rry += TWO_PI_F;
  float d0 = g[0] - roi[0], d1 = g[1] - roi[1], d2 = g[2] - roi[2];
  float c = cosf(rry), s = sinf(rry);
  float lx = d0 * c + d1 * s;
  float ly = -d0 * s + d1 * c;
  float h = fmodf(g[6] - rry, TWO_PI_F);
  if (h < 0.0f) h += TWO_PI_F;
  if (h > HALF_PI_F && h < 3.0f * HALF_PI_F) {
    h = fmodf(h + PI_F, TWO_PI_F);
    if (h < 0.0f) h += TWO_PI_F;
  }
  if (h > PI_F) h -= TWO_PI_F;
  h = fminf(fmaxf(h, -HALF_PI_F), HALF_PI_F);
  float* o = out + (size_t)(BB * POST * 7 + 2 * BB * POST) + (size_t)idx * 8;
  o[0] = lx; o[1] = ly; o[2] = d2;
  o[3] = g[3]; o[4] = g[4]; o[5] = g[5];
  o[6] = h; o[7] = g[7];
}

// ---------------------------------------------------------------------------
extern "C" void kernel_launch(void* const* d_in, const int* in_sizes, int n_in,
                              void* d_out, int out_size, void* d_ws,
                              size_t ws_size, hipStream_t stream) {
  (void)in_sizes; (void)n_in; (void)out_size; (void)ws_size;
  const float* boxes = (const float*)d_in[0];   // (B, N, 7)
  const float* cls   = (const float*)d_in[1];   // (B, N, 3)
  const float* gt    = (const float*)d_in[2];   // (B, POST, 8)
  float* out = (float*)d_out;

  // workspace layout (float units)
  float* ws       = (float*)d_ws;
  float* scores   = ws;                                    // B*N = 65536
  int*   hist     = (int*)(ws + (size_t)BB * NN);          // B*NBIN = 262144
  int*   thr_bin  = hist + (size_t)BB * NBIN;              // B
  int*   cand_cnt = thr_bin + BB;                          // B
  float* candS    = (float*)(cand_cnt + BB);               // B*MAXC = 16384
  int*   candI    = (int*)(candS + (size_t)BB * MAXC);     // B*MAXC = 16384
  float* top_s    = (float*)(candI + (size_t)BB * MAXC);   // B*PRE  = 2048
  int*   top_i    = (int*)(top_s + (size_t)BB * PRE);      // B*PRE  = 2048
  float* boxesG   = (float*)top_i + (size_t)BB * PRE;      // B*PRE*7 = 14336
  float* norms    = boxesG + (size_t)BB * PRE * 7;         // B*PRE
  float* rads     = norms + (size_t)BB * PRE;              // B*PRE
  float* gate     = rads + (size_t)BB * PRE;               // B*PRE*PRE
  float* iou      = gate + (size_t)BB * PRE * PRE;         // B*PRE*PRE

  int nzero = BB * NBIN + 2 * BB;   // hist + thr_bin + cand_cnt (contiguous)

  score_kernel   <<<(BB * NN) / 256, 256, 0, stream>>>(cls, scores);
  zero_kernel    <<<(nzero + 255) / 256, 256, 0, stream>>>(hist, nzero);
  hist_count_kernel<<<(BB * NN) / 256, 256, 0, stream>>>(scores, hist);
  hist_scan_kernel<<<BB, 1024, 0, stream>>>(hist, thr_bin);
  compact_kernel <<<(BB * NN) / 256, 256, 0, stream>>>(scores, thr_bin,
                                                       cand_cnt, candS, candI);
  cand_rank_kernel<<<BB, 256, 0, stream>>>(candS, candI, cand_cnt,
                                           top_s, top_i);
  gather_kernel  <<<(BB * PRE) / 256, 256, 0, stream>>>(boxes, top_i, boxesG,
                                                        norms, rads);
  wmma_gate_kernel<<<BB * 32 * 32, 32, 0, stream>>>(boxesG, norms, rads, gate);
  iou_kernel     <<<(BB * PRE * PRE) / 256, 256, 0, stream>>>(boxesG, gate, iou);
  nms_select_kernel<<<BB, PRE, 0, stream>>>(iou, top_s, top_i, boxesG, cls, out);
  gtlocal_kernel <<<(BB * POST) / 128, 128, 0, stream>>>(gt, out);
}
